// Mamba2Wrapper_26225070309879
// MI455X (gfx1250) — matrix operand order repaired and checked by reading, not compiled, not run on hardware
//
#include <hip/hip_runtime.h>

// ---------------- shapes ----------------
#define D_MODEL   1024
#define D_INNER   2048
#define D_STATE   128
#define NHEADS    32
#define HEADDIM   64
#define CONV_DIM  2304
#define D_IN_PROJ 4384
#define CHUNK     64
#define BATCH     4
#define SEQLEN    2048
#define ROWS      (BATCH * SEQLEN)      // 8192
#define NCHUNK    (SEQLEN / CHUNK)      // 32
#define EPS       1e-5f

typedef __attribute__((ext_vector_type(16))) __bf16 v16bf;
typedef __attribute__((ext_vector_type(8)))  float  v8f;

// ---------------- helpers ----------------
__device__ __forceinline__ unsigned short f2bf(float f) {
  union { float f; unsigned int u; } v; v.f = f;
  unsigned int r = v.u + 0x7FFFu + ((v.u >> 16) & 1u);
  return (unsigned short)(r >> 16);
}

__device__ __forceinline__ v8f wmma_bf16(v16bf a, v16bf b, v8f c) {
  return __builtin_amdgcn_wmma_f32_16x16x32_bf16(false, a, false, b, (short)0, c,
                                                 false, false);
}

// A fragment (16x32 bf16). Source row-major, row = matrix row, ld in elements.
// Lane half h holds K = 16h..16h+15 contiguous.
__device__ __forceinline__ v16bf ld_frag_a(const unsigned short* base, int ld) {
  int lane = threadIdx.x & 31;
  int r = lane & 15;
  int kh = (lane >> 4) * 16;
  const unsigned short* p = base + (size_t)r * ld + kh;
  union { v16bf v; uint4 q[2]; } u;
  u.q[0] = *(const uint4*)(p);
  u.q[1] = *(const uint4*)(p + 8);
  return u.v;
}

// B fragment (32x16 bf16). Source stored N-major (row = output column), ld elems.
// Lane half h holds K = 16h..16h+15 contiguous.
__device__ __forceinline__ v16bf ld_frag_b(const unsigned short* base, int ld) {
  int lane = threadIdx.x & 31;
  int r = lane & 15;
  int kh = (lane >> 4) * 16;
  const unsigned short* p = base + (size_t)r * ld + kh;
  union { v16bf v; uint4 q[2]; } u;
  u.q[0] = *(const uint4*)(p);
  u.q[1] = *(const uint4*)(p + 8);
  return u.v;
}

// B fragment from f32 source (converted on the fly).
__device__ __forceinline__ v16bf ld_frag_b_f32(const float* base, int ld) {
  int lane = threadIdx.x & 31;
  int r = lane & 15;
  int kh = (lane >> 4) * 16;
  const float* p = base + (size_t)r * ld + kh;
  union { v16bf v; unsigned short s[16]; } u;
#pragma unroll
  for (int i = 0; i < 16; ++i) u.s[i] = f2bf(p[i]);
  return u.v;
}

// Store 16x16 f32 accumulator. dst row-major, ld elems.
__device__ __forceinline__ void st_acc(float* dst, int ld, v8f a) {
  int lane = threadIdx.x & 31;
  int n  = lane & 15;
  int mb = (lane >> 4) * 8;
#pragma unroll
  for (int j = 0; j < 8; ++j) dst[(size_t)(mb + j) * ld + n] = a[j];
}

// ---------------- conversion kernels ----------------
__global__ void __launch_bounds__(256)
cvt_bf16(const float* __restrict__ src, unsigned short* __restrict__ dst, int n) {
  int i = blockIdx.x * 256 + threadIdx.x;
  if (i < n) dst[i] = f2bf(src[i]);
}

// dst[n*rows + k] = src[k*cols + n]   (rows = K, cols = N)
__global__ void __launch_bounds__(256)
cvt_transpose_bf16(const float* __restrict__ src, unsigned short* __restrict__ dst,
                   int rows, int cols) {
  int i = blockIdx.x * 256 + threadIdx.x;
  if (i < rows * cols) {
    int n = i / rows, k = i % rows;
    dst[i] = f2bf(src[(size_t)k * cols + n]);
  }
}

// ---------------- GEMM: C(MxN) = A(MxK,bf16) * B(KxN) with B stored N-major ----
// Wave tile: 64 rows x (16*CT) cols. Block = 4 waves stacked in rows.
template <int CT>
__global__ void __launch_bounds__(128)
gemm_bf16(const unsigned short* __restrict__ A, const unsigned short* __restrict__ BT,
          float* __restrict__ C, int M, int N, int K) {
  int wave = threadIdx.x >> 5;
  int row0 = (blockIdx.y * 4 + wave) * 64;
  int col0 = blockIdx.x * (16 * CT);
  v8f acc[4][CT];
#pragma unroll
  for (int r = 0; r < 4; ++r)
#pragma unroll
    for (int c = 0; c < CT; ++c) acc[r][c] = (v8f)0.0f;

  for (int k = 0; k < K; k += 32) {
    v16bf af[4], bfm[CT];
#pragma unroll
    for (int r = 0; r < 4; ++r)
      af[r] = ld_frag_a(A + (size_t)(row0 + r * 16) * K + k, K);
#pragma unroll
    for (int c = 0; c < CT; ++c)
      bfm[c] = ld_frag_b(BT + (size_t)(col0 + c * 16) * K + k, K);
#pragma unroll
    for (int r = 0; r < 4; ++r)
#pragma unroll
      for (int c = 0; c < CT; ++c)
        acc[r][c] = wmma_bf16(af[r], bfm[c], acc[r][c]);
  }
#pragma unroll
  for (int r = 0; r < 4; ++r)
#pragma unroll
    for (int c = 0; c < CT; ++c)
      st_acc(C + (size_t)(row0 + r * 16) * N + col0 + c * 16, N, acc[r][c]);
}

// ---------------- depthwise causal conv (k=4) + SiLU ----------------
__global__ void __launch_bounds__(256)
conv_silu(const float* __restrict__ zx, const float* __restrict__ w,
          const float* __restrict__ b, float* __restrict__ xconv) {
  size_t idx = (size_t)blockIdx.x * 256 + threadIdx.x;
  if (idx >= (size_t)ROWS * CONV_DIM) return;
  int c = (int)(idx % CONV_DIM);
  size_t row = idx / CONV_DIM;
  int l = (int)(row % SEQLEN);
  float acc = b[c];
#pragma unroll
  for (int j = 0; j < 4; ++j) {
    int ls = l - 3 + j;
    if (ls >= 0)
      acc += zx[(row + (size_t)(ls - l)) * D_IN_PROJ + D_INNER + c] * w[c * 4 + j];
  }
  xconv[row * CONV_DIM + c] = acc / (1.f + expf(-acc));
}

// ---------------- dt = softplus(raw + bias), dtA = dt * (-exp(A_log)) -------
__global__ void __launch_bounds__(256)
dt_kernel(const float* __restrict__ zx, const float* __restrict__ dtb,
          const float* __restrict__ Alog, float* __restrict__ dt,
          float* __restrict__ dtA) {
  int idx = blockIdx.x * 256 + threadIdx.x;
  if (idx >= ROWS * NHEADS) return;
  int h = idx & 31;
  size_t row = (size_t)idx >> 5;
  float x = zx[row * D_IN_PROJ + (D_INNER + CONV_DIM) + h] + dtb[h];
  float sp = (x > 20.f) ? x : log1pf(expf(x));
  dt[idx]  = sp;
  dtA[idx] = sp * (-expf(Alog[h]));
}

// ---------------- chunked SSD (one workgroup per (batch, head)) -------------
__global__ void __launch_bounds__(128)
ssd_kernel(const float* __restrict__ xconv, const float* __restrict__ dt,
           const float* __restrict__ dtA, float* __restrict__ Y) {
  const int b = blockIdx.x >> 5;
  const int h = blockIdx.x & 31;
  const int tid = threadIdx.x;
  const int wave = tid >> 5;
  const int lane = tid & 31;
  const int n16 = lane & 15;
  const int mb  = (lane >> 4) * 8;

  __shared__ float         sPrev[HEADDIM * D_STATE];           // 64x128 f32
  __shared__ unsigned short sC  [CHUNK * D_STATE];             // C  (l,n)
  __shared__ unsigned short sCe [CHUNK * D_STATE];             // C*exp(Acum)
  __shared__ unsigned short sB  [CHUNK * D_STATE];             // B  (l,n)
  __shared__ unsigned short sBT [D_STATE * CHUNK];             // B^T (n,l)
  __shared__ unsigned short sXT [HEADDIM * CHUNK];             // (x*dt)^T (p,l)
  __shared__ unsigned short sXdT[HEADDIM * CHUNK];             // with decay
  __shared__ unsigned short sM  [CHUNK * CHUNK];               // masked G
  __shared__ float sAcum[CHUNK], sDecay[CHUNK], sExpA[CHUNK];

  for (int i = tid; i < HEADDIM * D_STATE; i += 128) sPrev[i] = 0.f;
  __syncthreads();

  for (int c = 0; c < NCHUNK; ++c) {
    const int l0 = c * CHUNK;
    // ---- parallel stage of dt*a, then serial prefix in LDS ----
    if (tid < CHUNK)
      sAcum[tid] = dtA[((size_t)(b * SEQLEN + l0 + tid)) * NHEADS + h];
    __syncthreads();
    if (tid == 0) {
      float s = 0.f;
      for (int l = 0; l < CHUNK; ++l) { s += sAcum[l]; sAcum[l] = s; }
    }
    __syncthreads();
    for (int l = tid; l < CHUNK; l += 128) {
      sDecay[l] = expf(sAcum[CHUNK - 1] - sAcum[l]);
      sExpA[l]  = expf(sAcum[l]);
    }
    __syncthreads();
    // ---- stage C, Cexp, B, B^T ----
    for (int i = tid; i < CHUNK * D_STATE; i += 128) {
      int l = i >> 7, n = i & 127;
      size_t row = (size_t)(b * SEQLEN + l0 + l);
      float Cv = xconv[row * CONV_DIM + D_INNER + D_STATE + n];
      float Bv = xconv[row * CONV_DIM + D_INNER + n];
      sC[i]  = f2bf(Cv);
      sCe[i] = f2bf(Cv * sExpA[l]);
      sB[i]  = f2bf(Bv);
      sBT[n * CHUNK + l] = f2bf(Bv);
    }
    // ---- stage X^T (dt-weighted, and with state decay) ----
    for (int i = tid; i < HEADDIM * CHUNK; i += 128) {
      int p = i >> 6, l = i & 63;
      size_t row = (size_t)(b * SEQLEN + l0 + l);
      float xw = xconv[row * CONV_DIM + h * HEADDIM + p] * dt[row * NHEADS + h];
      sXT[i]  = f2bf(xw);
      sXdT[i] = f2bf(xw * sDecay[l]);
    }
    __syncthreads();

    // ---- G = C * B^T (64x64, K=128), mask with L, store bf16 into sM ----
    {
      v8f acc[4];
#pragma unroll
      for (int t = 0; t < 4; ++t) acc[t] = (v8f)0.0f;
      for (int ks = 0; ks < D_STATE; ks += 32) {
        v16bf af = ld_frag_a(sC + wave * 16 * D_STATE + ks, D_STATE);
#pragma unroll
        for (int t = 0; t < 4; ++t) {
          v16bf bf_ = ld_frag_b(sB + t * 16 * D_STATE + ks, D_STATE);
          acc[t] = wmma_bf16(af, bf_, acc[t]);
        }
      }
#pragma unroll
      for (int t = 0; t < 4; ++t)
#pragma unroll
        for (int j = 0; j < 8; ++j) {
          int ig = wave * 16 + mb + j;
          int jg = t * 16 + n16;
          float v = (ig >= jg) ? acc[t][j] * expf(sAcum[ig] - sAcum[jg]) : 0.f;
          sM[ig * CHUNK + jg] = f2bf(v);
        }
    }
    __syncthreads();

    // ---- Y = M*X  +  Cexp*prev^T (64x64 rows l, cols p) ----
    v8f accY[4];
#pragma unroll
    for (int t = 0; t < 4; ++t) accY[t] = (v8f)0.0f;
    for (int ks = 0; ks < CHUNK; ks += 32) {                 // diag: K over l
      v16bf af = ld_frag_a(sM + wave * 16 * CHUNK + ks, CHUNK);
#pragma unroll
      for (int t = 0; t < 4; ++t) {
        v16bf bf_ = ld_frag_b(sXT + t * 16 * CHUNK + ks, CHUNK);
        accY[t] = wmma_bf16(af, bf_, accY[t]);
      }
    }
    for (int ks = 0; ks < D_STATE; ks += 32) {               // off: K over n
      v16bf af = ld_frag_a(sCe + wave * 16 * D_STATE + ks, D_STATE);
#pragma unroll
      for (int t = 0; t < 4; ++t) {
        v16bf bf_ = ld_frag_b_f32(sPrev + t * 16 * D_STATE + ks, D_STATE);
        accY[t] = wmma_bf16(af, bf_, accY[t]);
      }
    }
    __syncthreads();   // all sPrev reads complete before the state update

    // ---- chunk state S (64p x 128n, K over l) and prev update ----
    {
      v8f accS[8];
#pragma unroll
      for (int t = 0; t < 8; ++t) accS[t] = (v8f)0.0f;
      for (int ks = 0; ks < CHUNK; ks += 32) {
        v16bf af = ld_frag_a(sXdT + wave * 16 * CHUNK + ks, CHUNK);
#pragma unroll
        for (int t = 0; t < 8; ++t) {
          v16bf bf_ = ld_frag_b(sBT + t * 16 * CHUNK + ks, CHUNK);
          accS[t] = wmma_bf16(af, bf_, accS[t]);
        }
      }
      float asum = expf(sAcum[CHUNK - 1]);
#pragma unroll
      for (int t = 0; t < 8; ++t)
#pragma unroll
        for (int j = 0; j < 8; ++j) {
          int p = wave * 16 + mb + j;
          int n = t * 16 + n16;
          sPrev[p * D_STATE + n] = sPrev[p * D_STATE + n] * asum + accS[t][j];
        }
    }
    // ---- write Y ----
#pragma unroll
    for (int t = 0; t < 4; ++t)
#pragma unroll
      for (int j = 0; j < 8; ++j) {
        int l = wave * 16 + mb + j;
        int p = t * 16 + n16;
        Y[((size_t)(b * SEQLEN + l0 + l)) * D_INNER + h * HEADDIM + p] = accY[t][j];
      }
    __syncthreads();   // protect LDS staging for next chunk
  }
}

// ---------------- skip + gate + RMSNorm -> bf16 ----------------
__global__ void __launch_bounds__(256)
gate_rms(const float* __restrict__ Y, const float* __restrict__ xconv,
         const float* __restrict__ zx, const float* __restrict__ Dv,
         const float* __restrict__ rmsw, unsigned short* __restrict__ ynorm) {
  size_t row = blockIdx.x;
  __shared__ float red[256];
  float loc[8];
  float ss = 0.f;
#pragma unroll
  for (int i = 0; i < 8; ++i) {
    int d = threadIdx.x + i * 256;
    int h = d >> 6;
    float yv = Y[row * D_INNER + d] + xconv[row * CONV_DIM + d] * Dv[h];
    float z  = zx[row * D_IN_PROJ + d];
    yv *= z / (1.f + expf(-z));
    loc[i] = yv;
    ss += yv * yv;
  }
  red[threadIdx.x] = ss;
  __syncthreads();
  for (int s = 128; s > 0; s >>= 1) {
    if (threadIdx.x < (unsigned)s) red[threadIdx.x] += red[threadIdx.x + s];
    __syncthreads();
  }
  float inv = rsqrtf(red[0] / (float)D_INNER + EPS);
#pragma unroll
  for (int i = 0; i < 8; ++i) {
    int d = threadIdx.x + i * 256;
    ynorm[row * D_INNER + d] = f2bf(loc[i] * inv * rmsw[d]);
  }
}

// ---------------- residual + LayerNorm ----------------
__global__ void __launch_bounds__(256)
final_ln(const float* __restrict__ hs, const float* __restrict__ outmm,
         const float* __restrict__ g, const float* __restrict__ bt,
         float* __restrict__ out) {
  size_t row = blockIdx.x;
  __shared__ float r1[256], r2[256];
  float v[4];
  float s = 0.f, s2 = 0.f;
#pragma unroll
  for (int i = 0; i < 4; ++i) {
    int d = threadIdx.x + i * 256;
    float x = hs[row * D_MODEL + d] + outmm[row * D_MODEL + d];
    v[i] = x; s += x; s2 += x * x;
  }
  r1[threadIdx.x] = s; r2[threadIdx.x] = s2;
  __syncthreads();
  for (int st = 128; st > 0; st >>= 1) {
    if (threadIdx.x < (unsigned)st) {
      r1[threadIdx.x] += r1[threadIdx.x + st];
      r2[threadIdx.x] += r2[threadIdx.x + st];
    }
    __syncthreads();
  }
  float mu  = r1[0] / (float)D_MODEL;
  float var = r2[0] / (float)D_MODEL - mu * mu;
  float inv = rsqrtf(var + EPS);
#pragma unroll
  for (int i = 0; i < 4; ++i) {
    int d = threadIdx.x + i * 256;
    out[row * D_MODEL + d] = (v[i] - mu) * inv * g[d] + bt[d];
  }
}

// ---------------- host ----------------
extern "C" void kernel_launch(void* const* d_in, const int* in_sizes, int n_in,
                              void* d_out, int out_size, void* d_ws, size_t ws_size,
                              hipStream_t stream) {
  const float* hs    = (const float*)d_in[0];
  const float* Win   = (const float*)d_in[1];
  const float* convw = (const float*)d_in[2];
  const float* convb = (const float*)d_in[3];
  const float* dtb   = (const float*)d_in[4];
  const float* Alog  = (const float*)d_in[5];
  const float* Dv    = (const float*)d_in[6];
  const float* rmsw  = (const float*)d_in[7];
  const float* Wout  = (const float*)d_in[8];
  const float* lng   = (const float*)d_in[9];
  const float* lnb   = (const float*)d_in[10];
  float* out = (float*)d_out;

  char* ws = (char*)d_ws;
  auto take = [&](size_t bytes) {
    char* p = ws;
    ws += (bytes + 255) & ~(size_t)255;
    return p;
  };
  float*          zx    = (float*)take((size_t)ROWS * D_IN_PROJ * 4);
  float*          xconv = (float*)take((size_t)ROWS * CONV_DIM * 4);
  float*          dtv   = (float*)take((size_t)ROWS * NHEADS * 4);
  float*          dtA   = (float*)take((size_t)ROWS * NHEADS * 4);
  float*          Yb    = (float*)take((size_t)ROWS * D_INNER * 4);
  unsigned short* hsb   = (unsigned short*)take((size_t)ROWS * D_MODEL * 2);
  unsigned short* WinT  = (unsigned short*)take((size_t)D_IN_PROJ * D_MODEL * 2);
  unsigned short* ynorm = (unsigned short*)take((size_t)ROWS * D_INNER * 2);
  unsigned short* WoutT = (unsigned short*)take((size_t)D_MODEL * D_INNER * 2);
  float*          outmm = (float*)take((size_t)ROWS * D_MODEL * 4);

  // bf16 conversions
  {
    int n = ROWS * D_MODEL;
    cvt_bf16<<<(n + 255) / 256, 256, 0, stream>>>(hs, hsb, n);
  }
  {
    int n = D_IN_PROJ * D_MODEL;
    cvt_transpose_bf16<<<(n + 255) / 256, 256, 0, stream>>>(Win, WinT, D_MODEL, D_IN_PROJ);
  }
  {
    int n = D_MODEL * D_INNER;
    cvt_transpose_bf16<<<(n + 255) / 256, 256, 0, stream>>>(Wout, WoutT, D_INNER, D_MODEL);
  }

  // in_proj GEMM: (8192x1024) x (1024x4384), N = 32*137 -> 32-wide col tiles
  gemm_bf16<2><<<dim3(D_IN_PROJ / 32, ROWS / 256), 128, 0, stream>>>(
      hsb, WinT, zx, ROWS, D_IN_PROJ, D_MODEL);

  // conv + SiLU, dt
  {
    size_t n = (size_t)ROWS * CONV_DIM;
    conv_silu<<<(unsigned)((n + 255) / 256), 256, 0, stream>>>(zx, convw, convb, xconv);
  }
  {
    int n = ROWS * NHEADS;
    dt_kernel<<<(n + 255) / 256, 256, 0, stream>>>(zx, dtb, Alog, dtv, dtA);
  }

  // SSD
  ssd_kernel<<<BATCH * NHEADS, 128, 0, stream>>>(xconv, dtv, dtA, Yb);

  // gate + RMSNorm -> bf16
  gate_rms<<<ROWS, 256, 0, stream>>>(Yb, xconv, zx, Dv, rmsw, ynorm);

  // out_proj GEMM: (8192x2048) x (2048x1024), N divisible by 64 -> 64-wide tiles
  gemm_bf16<4><<<dim3(D_MODEL / 64, ROWS / 256), 128, 0, stream>>>(
      ynorm, WoutT, outmm, ROWS, D_MODEL, D_INNER);

  // residual + LayerNorm
  final_ln<<<ROWS, 256, 0, stream>>>(hs, outmm, lng, lnb, out);

  (void)in_sizes; (void)n_in; (void)out_size; (void)ws_size;
}
